// SAGPoolClassifier_7275674599564
// MI455X (gfx1250) — matrix operand order repaired and checked by reading, not compiled, not run on hardware
//
#include <hip/hip_runtime.h>
#include <hip/hip_bf16.h>

// SAGPool graph classifier for MI455X (gfx1250, wave32, WMMA + TDM).
// One workgroup (8 wave32) per graph; bf16 WMMA for both GraphConv GEMMs;
// adjacency strips DMA'd global->LDS by the Tensor Data Mover (double
// buffered, s_wait_tensorcnt); everything between HBM reads lives in LDS.

typedef __bf16 bf16_t;
typedef __attribute__((ext_vector_type(16))) __bf16 v16bf;
typedef __attribute__((ext_vector_type(8)))  float  v8f;
typedef __attribute__((ext_vector_type(4)))  unsigned int u32x4;
typedef __attribute__((ext_vector_type(8)))  int   i32x8;
typedef __attribute__((ext_vector_type(4)))  int   i32x4;

#define NGRAPHS 1024
#define DHID    256   // hidden width of every GraphConv layer

__device__ __forceinline__ float lrelu(float v) { return v > 0.f ? v : 0.01f * v; }

// ---- fragment-major LDS layouts (ISA 7.12.2, wave32) -----------------------
// A (16x32 bf16): lane l holds row m=l%16; element e -> K = (e/8)*16 + (l/16)*8 + e%8
__device__ __forceinline__ int a_dst(int m, int k, int K) {
  int R = m >> 4, S = k >> 5, kl = k & 31;
  int grp = kl >> 4, rem = kl & 15, half = rem >> 3, rr = rem & 7;
  int lane = (half << 4) + (m & 15);
  int e = (grp << 3) + rr;
  return ((R * (K >> 5) + S) * 32 + lane) * 16 + e;
}
// B (32x16 bf16): lane l holds col n=l%16; element e -> K = (e/8)*16 + (l/16)*8 + e%8
__device__ __forceinline__ int b_dst(int k, int n) {
  int S = k >> 5, kl = k & 31, half = (kl >> 3) & 1, e = ((kl >> 4) << 3) + (kl & 7), T = n >> 4;
  int lane = (half << 4) + (n & 15);
  return ((S * 16 + T) * 32 + lane) * 16 + e;
}

// Build an A fragment (16x32) from a row-major f32 tile (global or LDS).
__device__ __forceinline__ v16bf tile_frag(const float* base, int k0, int lane, int ld) {
  const float* r = base + (size_t)(lane & 15) * ld + k0 + ((lane >> 4) << 3);
  float4 c0 = *(const float4*)(r);
  float4 c1 = *(const float4*)(r + 4);
  float4 c2 = *(const float4*)(r + 16);
  float4 c3 = *(const float4*)(r + 20);
  v16bf f;
  f[0]=(bf16_t)c0.x; f[1]=(bf16_t)c0.y; f[2]=(bf16_t)c0.z; f[3]=(bf16_t)c0.w;
  f[4]=(bf16_t)c1.x; f[5]=(bf16_t)c1.y; f[6]=(bf16_t)c1.z; f[7]=(bf16_t)c1.w;
  f[8]=(bf16_t)c2.x; f[9]=(bf16_t)c2.y; f[10]=(bf16_t)c2.z; f[11]=(bf16_t)c2.w;
  f[12]=(bf16_t)c3.x; f[13]=(bf16_t)c3.y; f[14]=(bf16_t)c3.z; f[15]=(bf16_t)c3.w;
  return f;
}
// K=16 adjacency (layer 2): zero-pad K 16..31.
__device__ __forceinline__ v16bf tile_frag_pad16(const float* base, int lane) {
  const float* r = base + (size_t)(lane & 15) * 16 + ((lane >> 4) << 3);
  float4 c0 = *(const float4*)(r);
  float4 c1 = *(const float4*)(r + 4);
  v16bf f;
  f[0]=(bf16_t)c0.x; f[1]=(bf16_t)c0.y; f[2]=(bf16_t)c0.z; f[3]=(bf16_t)c0.w;
  f[4]=(bf16_t)c1.x; f[5]=(bf16_t)c1.y; f[6]=(bf16_t)c1.z; f[7]=(bf16_t)c1.w;
  bf16_t z = (bf16_t)0.f;
  f[8]=z; f[9]=z; f[10]=z; f[11]=z; f[12]=z; f[13]=z; f[14]=z; f[15]=z;
  return f;
}

__device__ __forceinline__ v8f wmma_bf16(v16bf a, v16bf b, v8f c) {
  return __builtin_amdgcn_wmma_f32_16x16x32_bf16(
      false, a, false, b, (short)0, c, false, false);
}

// ---- Tensor Data Mover: DMA a [16 x nodes] f32 tile (row-major) into LDS ---
// D# per CDNA5 ISA: group0 = {count/lds/global/type},
// group1 = {data_size, tensor dims, tile dims, dim0 stride}. 2D -> groups 2/3 zero.
__device__ __forceinline__ void tdm_load_rows16(const float* gsrc, unsigned lds_off,
                                                int nodes) {
  unsigned long long ga = (unsigned long long)(uintptr_t)gsrc;
  u32x4 g0;
  g0[0] = 1u;                                            // count=1, user mode
  g0[1] = lds_off;                                       // lds_addr (bytes)
  g0[2] = (unsigned)ga;                                  // global_addr[31:0]
  g0[3] = (unsigned)((ga >> 32) & 0x01ffffffu) | (2u << 30);  // addr[56:32]|type=2
  i32x8 g1;
  g1[0] = (int)(2u << 16);                               // wg_mask=0, data_size=4B
  g1[1] = (int)(((unsigned)nodes & 0xffffu) << 16);      // tensor_dim0 lo16
  g1[2] = (int)((((unsigned)nodes >> 16) & 0xffffu) |
                (((unsigned)nodes & 0xffffu) << 16));    // dim0 hi | tensor_dim1 lo
  g1[3] = (int)((((unsigned)nodes >> 16) & 0xffffu) |
                (((unsigned)nodes & 0xffffu) << 16));    // dim1 hi | tile_dim0
  g1[4] = 16;                                            // tile_dim1=16 rows
  g1[5] = nodes;                                         // tensor_dim0_stride lo32
  g1[6] = 0;
  g1[7] = 0;
  i32x4 gz = {0, 0, 0, 0};
#if __clang_major__ >= 23
  i32x8 gz8 = {0, 0, 0, 0, 0, 0, 0, 0};
  __builtin_amdgcn_tensor_load_to_lds(g0, g1, gz, gz, gz8, 0);
#else
  __builtin_amdgcn_tensor_load_to_lds(g0, g1, gz, gz, 0);
#endif
}

// ---------------------------------------------------------------------------
// One GraphConv + SAGPool layer, one graph per block, 256 threads (8 waves).
// ---------------------------------------------------------------------------
template <int NODES, int KIN, int KPOOL, bool WRITE_POOL>
__global__ void sag_layer_kernel(const float* __restrict__ xin,    // [G][NODES][KIN]
                                 const float* __restrict__ adjin,  // [G][NODES][NODES]
                                 const float* __restrict__ W,      // [KIN][DHID]
                                 const float* __restrict__ b,      // [DHID]
                                 const float* __restrict__ Ws,     // [DHID]
                                 const float* __restrict__ bs,     // [1]
                                 float* __restrict__ xout,         // [G][KPOOL][DHID]
                                 float* __restrict__ aout,         // [G][KPOOL][KPOOL]
                                 float* __restrict__ ro,           // [G][1536]
                                 int ro_off) {
  constexpr int KS1 = KIN / 32;                       // K-steps of GEMM1
  constexpr int KS2 = (NODES >= 32) ? NODES / 32 : 1; // K-steps of GEMM2
  constexpr int STRIPS = NODES / 16;
  constexpr int HF_EL = KS2 * 16 * 512;               // H as B-fragments

  extern __shared__ char smem[];
  float* normS  = (float*)smem;                       // [NODES]
  float* t1     = normS + NODES;                      // [NODES]
  float* score  = t1 + NODES;                         // [NODES]
  int*   selidx = (int*)(score + NODES);              // [KPOOL]
  float* selval = (float*)(selidx + KPOOL);           // [KPOOL]
  bf16_t* XF = (bf16_t*)(smem + 4096);                // [NODES*KIN]  A-frags of X*norm
  bf16_t* WF = XF + NODES * KIN;                      // [KIN*DHID]   B-frags of W
  bf16_t* HF = WF + KIN * DHID;                       // [HF_EL]      B-frags of H
  float* ADJS = (float*)(HF + HF_EL);                 // [2][16][NODES] TDM staging
  bf16_t* XO = XF;                                    // [NODES*DHID] activations (reuse)

  const int g = blockIdx.x;
  const int tid = threadIdx.x;
  const int lane = tid & 31, wave = tid >> 5;
  const float* adjg = adjin + (size_t)g * NODES * NODES;
  const float* xg   = xin   + (size_t)g * NODES * KIN;
  const unsigned adjs_base = (unsigned)((char*)ADJS - smem);

  // --- degrees -> D^{-1/2} --------------------------------------------------
  if (tid < NODES) {
    float d = 0.f;
    const float4* r = (const float4*)(adjg + (size_t)tid * NODES);
#pragma unroll 4
    for (int i = 0; i < NODES / 4; ++i) { float4 v = r[i]; d += v.x + v.y + v.z + v.w; }
    normS[tid] = rsqrtf(fmaxf(d, 1.f));
  }
  if (NODES == 16) {  // zero the padded half of the H fragments (K 16..31)
    for (int i = tid; i < HF_EL; i += blockDim.x) HF[i] = (bf16_t)0.f;
  }
  __syncthreads();

  // --- stage A-frags of (X * norm) and B-frags of W in bf16 -----------------
  for (int i = tid; i < NODES * KIN; i += blockDim.x) {
    int m = i / KIN, k = i - m * KIN;
    XF[a_dst(m, k, KIN)] = (bf16_t)(xg[i] * normS[m]);
  }
  for (int i = tid; i < KIN * DHID; i += blockDim.x) {
    int k = i >> 8, n = i & 255;
    WF[b_dst(k, n)] = (bf16_t)W[i];
  }
  __syncthreads();

  // --- GEMM1: H = (X*norm) @ W -> HF; two independent WMMA chains per wave --
  for (int R = 0; R < STRIPS; ++R) {
    v16bf afr[KS1];
#pragma unroll
    for (int S = 0; S < KS1; ++S)
      afr[S] = *(const v16bf*)(XF + ((R * KS1 + S) * 32 + lane) * 16);
    const int T0 = wave, T1 = wave + 8;
    v8f acc0 = {0.f,0.f,0.f,0.f,0.f,0.f,0.f,0.f};
    v8f acc1 = acc0;
#pragma unroll
    for (int S = 0; S < KS1; ++S) {
      v16bf b0 = *(const v16bf*)(WF + ((S * 16 + T0) * 32 + lane) * 16);
      v16bf b1 = *(const v16bf*)(WF + ((S * 16 + T1) * 32 + lane) * 16);
      acc0 = wmma_bf16(afr[S], b0, acc0);
      acc1 = wmma_bf16(afr[S], b1, acc1);
    }
#pragma unroll
    for (int v = 0; v < 8; ++v) {
      int mg = R * 16 + v + ((lane >> 4) << 3);
      int nl = lane & 15;
      HF[b_dst(mg, T0 * 16 + nl)] = (bf16_t)acc0[v];
      HF[b_dst(mg, T1 * 16 + nl)] = (bf16_t)acc1[v];
    }
  }

  // --- GEMM2: Xout = lrelu( (A @ H) * norm + b ) ----------------------------
  // Adjacency strips (16 rows x NODES) are DMA'd to LDS by the TDM,
  // double-buffered and overlapped with the WMMA work of the previous strip.
  if (wave == 0)
    tdm_load_rows16(adjg, adjs_base, NODES);  // strip 0
  for (int R = 0; R < STRIPS; ++R) {
    if (wave == 0) __builtin_amdgcn_s_wait_tensorcnt(0);
    __syncthreads();  // strip R staged (and HF complete on first iteration)
    if (wave == 0 && R + 1 < STRIPS)
      tdm_load_rows16(adjg + (size_t)(R + 1) * 16 * NODES,
                      adjs_base + (unsigned)(((R + 1) & 1) * 16 * NODES * 4), NODES);
    const float* atile = ADJS + (R & 1) * 16 * NODES;

    v16bf afr[KS2];
#pragma unroll
    for (int S = 0; S < KS2; ++S) {
      if constexpr (NODES == 16)
        afr[S] = tile_frag_pad16(atile, lane);
      else
        afr[S] = tile_frag(atile, S * 32, lane, NODES);
    }
    const int T0 = wave, T1 = wave + 8;
    v8f acc0 = {0.f,0.f,0.f,0.f,0.f,0.f,0.f,0.f};
    v8f acc1 = acc0;
#pragma unroll
    for (int S = 0; S < KS2; ++S) {
      v16bf b0 = *(const v16bf*)(HF + ((S * 16 + T0) * 32 + lane) * 16);
      v16bf b1 = *(const v16bf*)(HF + ((S * 16 + T1) * 32 + lane) * 16);
      acc0 = wmma_bf16(afr[S], b0, acc0);
      acc1 = wmma_bf16(afr[S], b1, acc1);
    }
#pragma unroll
    for (int v = 0; v < 8; ++v) {
      int mg = R * 16 + v + ((lane >> 4) << 3);
      int nl = lane & 15;
      float v0 = acc0[v] * normS[mg] + b[T0 * 16 + nl];
      float v1 = acc1[v] * normS[mg] + b[T1 * 16 + nl];
      XO[mg * DHID + T0 * 16 + nl] = (bf16_t)lrelu(v0);
      XO[mg * DHID + T1 * 16 + nl] = (bf16_t)lrelu(v1);
    }
  }
  __syncthreads();

  // --- SAGPool score: s = norm * (A @ (norm * X @ Ws)) + bs -----------------
  if (tid < NODES) {
    float s = 0.f;
#pragma unroll 8
    for (int d = 0; d < DHID; ++d) s += (float)XO[tid * DHID + d] * Ws[d];
    t1[tid] = s * normS[tid];
  }
  __syncthreads();
  if (tid < NODES) {
    float sc = 0.f;
    const float* ar = adjg + (size_t)tid * NODES;
    for (int m = 0; m < NODES; ++m) sc += ar[m] * t1[m];
    score[tid] = sc * normS[tid] + bs[0];
  }
  __syncthreads();
  // top-KPOOL by exact rank (ties -> lower index first, matching lax.top_k)
  if (tid < NODES) {
    float my = score[tid];
    int rank = 0;
    for (int m = 0; m < NODES; ++m) {
      float o = score[m];
      rank += (o > my) || (o == my && m < tid);
    }
    if (rank < KPOOL) { selidx[rank] = tid; selval[rank] = my; }
  }
  __syncthreads();

  // --- pooled outputs + sum/max readout -------------------------------------
  if (WRITE_POOL) {
    float* xg_out = xout + (size_t)g * KPOOL * DHID;
    for (int i = tid; i < KPOOL * DHID; i += blockDim.x) {
      int j = i >> 8, c = i & 255;
      xg_out[i] = (float)XO[selidx[j] * DHID + c] * tanhf(selval[j]);
    }
    float* ag_out = aout + (size_t)g * KPOOL * KPOOL;
    for (int i = tid; i < KPOOL * KPOOL; i += blockDim.x) {
      int j = i / KPOOL, c = i - j * KPOOL;
      ag_out[i] = adjg[(size_t)selidx[j] * NODES + selidx[c]];
    }
  }
  {
    int c = tid;  // 256 threads == DHID columns
    float s = 0.f, mx = -3.4e38f;
#pragma unroll
    for (int j = 0; j < KPOOL; ++j) {
      float v = (float)XO[selidx[j] * DHID + c] * tanhf(selval[j]);
      s += v; mx = fmaxf(mx, v);
    }
    ro[(size_t)g * 1536 + ro_off + c]       = s;
    ro[(size_t)g * 1536 + ro_off + 256 + c] = mx;
  }
}

// ---------------------------------------------------------------------------
// Final MLP: out = sigmoid(lrelu(merged @ Wd1 + bd1) @ Wd2 + bd2)
// ---------------------------------------------------------------------------
__global__ void sag_mlp_kernel(const float* __restrict__ ro,   // [G][1536]
                               const float* __restrict__ Wd1,  // [1536][128]
                               const float* __restrict__ bd1,  // [128]
                               const float* __restrict__ Wd2,  // [128][2]
                               const float* __restrict__ bd2,  // [2]
                               float* __restrict__ out) {      // [G][2]
  extern __shared__ char smem[];
  float* mS = (float*)smem;        // [16*1536]
  float* hS = mS + 16 * 1536;      // [16*128]
  const int g0 = blockIdx.x * 16;
  const int tid = threadIdx.x;

  for (int i = tid; i < 16 * 1536; i += blockDim.x)
    mS[i] = ro[(size_t)g0 * 1536 + i];
  __syncthreads();

  for (int i = tid; i < 16 * 128; i += blockDim.x) {
    int gg = i >> 7, d = i & 127;
    float acc = bd1[d];
    const float* m = mS + gg * 1536;
#pragma unroll 8
    for (int k = 0; k < 1536; ++k) acc += m[k] * Wd1[k * 128 + d];
    hS[i] = lrelu(acc);
  }
  __syncthreads();

  for (int i = tid; i < 32; i += blockDim.x) {
    int gg = i >> 1, o = i & 1;
    float acc = bd2[o];
    const float* h = hS + gg * 128;
#pragma unroll
    for (int d = 0; d < 128; ++d) acc += h[d] * Wd2[d * 2 + o];
    out[(size_t)(g0 + gg) * 2 + o] = 1.f / (1.f + expf(-acc));
  }
}

static constexpr size_t layer_shmem(int NODES, int KIN) {
  size_t ks2 = (NODES >= 32) ? (size_t)NODES / 32 : 1;
  return 4096 + 2ull * ((size_t)NODES * KIN + (size_t)KIN * DHID + ks2 * 16 * 512)
       + 2ull * 16 * NODES * 4;  // TDM double buffer
}

extern "C" void kernel_launch(void* const* d_in, const int* in_sizes, int n_in,
                              void* d_out, int out_size, void* d_ws, size_t ws_size,
                              hipStream_t stream) {
  const float* feat = (const float*)d_in[0];
  const float* adj  = (const float*)d_in[1];
  const float* W0  = (const float*)d_in[2];  const float* b0  = (const float*)d_in[3];
  const float* Ws0 = (const float*)d_in[4];  const float* bs0 = (const float*)d_in[5];
  const float* W1  = (const float*)d_in[6];  const float* b1  = (const float*)d_in[7];
  const float* Ws1 = (const float*)d_in[8];  const float* bs1 = (const float*)d_in[9];
  const float* W2  = (const float*)d_in[10]; const float* b2  = (const float*)d_in[11];
  const float* Ws2 = (const float*)d_in[12]; const float* bs2 = (const float*)d_in[13];
  const float* Wd1 = (const float*)d_in[14]; const float* bd1 = (const float*)d_in[15];
  const float* Wd2 = (const float*)d_in[16]; const float* bd2 = (const float*)d_in[17];
  float* out = (float*)d_out;

  char* ws = (char*)d_ws;
  size_t off = 0;
  float* x1 = (float*)(ws + off); off += (size_t)NGRAPHS * 32 * 256 * sizeof(float);
  float* a1 = (float*)(ws + off); off += (size_t)NGRAPHS * 32 * 32  * sizeof(float);
  float* x2 = (float*)(ws + off); off += (size_t)NGRAPHS * 16 * 256 * sizeof(float);
  float* a2 = (float*)(ws + off); off += (size_t)NGRAPHS * 16 * 16  * sizeof(float);
  float* ro = (float*)(ws + off); off += (size_t)NGRAPHS * 1536     * sizeof(float);

  sag_layer_kernel<256, 128, 32, true>
      <<<NGRAPHS, 256, layer_shmem(256, 128), stream>>>(
          feat, adj, W0, b0, Ws0, bs0, x1, a1, ro, 0);
  sag_layer_kernel<32, 256, 16, true>
      <<<NGRAPHS, 256, layer_shmem(32, 256), stream>>>(
          x1, a1, W1, b1, Ws1, bs1, x2, a2, ro, 512);
  sag_layer_kernel<16, 256, 8, false>
      <<<NGRAPHS, 256, layer_shmem(16, 256), stream>>>(
          x2, a2, W2, b2, Ws2, bs2, nullptr, nullptr, ro, 1024);
  sag_mlp_kernel<<<NGRAPHS / 16, 256, (16 * 1536 + 16 * 128) * sizeof(float), stream>>>(
      ro, Wd1, bd1, Wd2, bd2, out);
}
